// STWithRSbySPP_DGL_37941741093505
// MI455X (gfx1250) — compile-verified, matrix operand order checked by reading, not run on hardware
//
#include <hip/hip_runtime.h>
#include <hip/hip_bf16.h>

typedef __attribute__((ext_vector_type(16))) _Float16 v16h;
typedef __attribute__((ext_vector_type(8)))  float    v8f;

#define DEV __device__ __forceinline__

// ---- WMMA helpers (CDNA5 wave32, 16x16x32 f16 -> f32) -----------------------
// A (16x32) and B(32x16 = W[n][k]) fragments share the same lane mapping:
//   lane -> row/col = lane&15 ; K base = (lane>>4)*8, elements {kb..kb+7, kb+16..kb+23}
DEV v16h frag_ld(const _Float16* p, int ld) {
  int lane = threadIdx.x & 31;
  const _Float16* q = p + (size_t)(lane & 15) * ld + ((lane >> 4) << 3);
  v16h v;
#pragma unroll
  for (int j = 0; j < 8; ++j) { v[j] = q[j]; v[8 + j] = q[16 + j]; }
  return v;
}

DEV void frag_st(float* p, int ld, v8f c) {
  int lane = threadIdx.x & 31;
  int col = lane & 15, rb = (lane >> 4) << 3;
#pragma unroll
  for (int v = 0; v < 8; ++v) p[(size_t)(rb + v) * ld + col] = c[v];
}

DEV v8f wmma16(v16h a, v16h b, v8f c) {
  return __builtin_amdgcn_wmma_f32_16x16x32_f16(false, a, false, b, (short)0, c,
                                                false, false);
}

DEV float sigmoidf(float x) { return 1.f / (1.f + expf(-x)); }

// ---- f32 -> f16 pack with row/K zero padding --------------------------------
__global__ void pack_f16(const float* __restrict__ src, _Float16* __restrict__ dst,
                         int rows, int rowsP, int K, int Kp) {
  size_t total = (size_t)rowsP * Kp;
  for (size_t i = (size_t)blockIdx.x * blockDim.x + threadIdx.x; i < total;
       i += (size_t)gridDim.x * blockDim.x) {
    size_t r = i / Kp; int c = (int)(i % Kp);
    float v = ((int)r < rows && c < K) ? src[r * K + c] : 0.f;
    dst[i] = (_Float16)v;
  }
}

// ---- Sentence BiLSTM, fused input+recurrent GEMM ----------------------------
// grid (128 row-tiles, 2 dirs), block 128 (4 waves; wave w == gate block w).
// X: [2048][50][224] f16 (K padded 200->224). Outputs tanh(mean_t h) -> sentpres.
__global__ void __launch_bounds__(128)
sent_lstm(const _Float16* __restrict__ X,
          const _Float16* __restrict__ Wih0, const _Float16* __restrict__ Wih1,
          const _Float16* __restrict__ Whh0, const _Float16* __restrict__ Whh1,
          const float* __restrict__ b0, const float* __restrict__ b1,
          float* __restrict__ sp32, _Float16* __restrict__ sp16) {
  __shared__ float    gbuf[16][512];
  __shared__ _Float16 hbuf[16][128];
  const int dir = blockIdx.y;
  const int rowbase = blockIdx.x * 16;
  const _Float16* Wih = dir ? Wih1 : Wih0;
  const _Float16* Whh = dir ? Whh1 : Whh0;
  const float* bias = dir ? b1 : b0;
  const int wave = threadIdx.x >> 5;
  const int tid = threadIdx.x;

  for (int i = tid; i < 16 * 128; i += 128) (&hbuf[0][0])[i] = (_Float16)0.f;
  float c_state[16], hsum[16];
#pragma unroll
  for (int i = 0; i < 16; ++i) { c_state[i] = 0.f; hsum[i] = 0.f; }
  __syncthreads();

  for (int t = 0; t < 50; ++t) {
    const int tt = dir ? (49 - t) : t;
    v8f acc[8];
#pragma unroll
    for (int n = 0; n < 8; ++n) acc[n] = (v8f){};
    const _Float16* Xp = X + ((size_t)rowbase * 50 + tt) * 224;
#pragma unroll
    for (int kt = 0; kt < 7; ++kt) {                       // x @ W_ih^T
      v16h a = frag_ld(Xp + kt * 32, 50 * 224);
#pragma unroll
      for (int n = 0; n < 8; ++n) {
        v16h b = frag_ld(Wih + (size_t)(wave * 128 + n * 16) * 224 + kt * 32, 224);
        acc[n] = wmma16(a, b, acc[n]);
      }
    }
#pragma unroll
    for (int kt = 0; kt < 4; ++kt) {                       // h @ W_hh^T
      v16h a = frag_ld(&hbuf[0][kt * 32], 128);
#pragma unroll
      for (int n = 0; n < 8; ++n) {
        v16h b = frag_ld(Whh + (size_t)(wave * 128 + n * 16) * 128 + kt * 32, 128);
        acc[n] = wmma16(a, b, acc[n]);
      }
    }
#pragma unroll
    for (int n = 0; n < 8; ++n) frag_st(&gbuf[0][wave * 128 + n * 16], 512, acc[n]);
    __syncthreads();
#pragma unroll
    for (int r = 0; r < 16; ++r) {                         // cell update, j = tid
      float gi = gbuf[r][tid]       + bias[tid];
      float gf = gbuf[r][128 + tid] + bias[128 + tid];
      float gg = gbuf[r][256 + tid] + bias[256 + tid];
      float go = gbuf[r][384 + tid] + bias[384 + tid];
      float c = sigmoidf(gf) * c_state[r] + sigmoidf(gi) * tanhf(gg);
      float h = sigmoidf(go) * tanhf(c);
      c_state[r] = c;
      hsum[r] += h;
      hbuf[r][tid] = (_Float16)h;
    }
    __syncthreads();
  }
#pragma unroll
  for (int r = 0; r < 16; ++r) {
    float v = tanhf(hsum[r] * (1.f / 50.f));
    size_t o = (size_t)(rowbase + r) * 256 + (size_t)dir * 128 + tid;
    sp32[o] = v;
    sp16[o] = (_Float16)v;
  }
}

// ---- Tag BiLSTM: batch 32 docs, seq = 64 sentences, in 256 ------------------
__global__ void __launch_bounds__(128)
tag_lstm(const _Float16* __restrict__ SP,     // [32*64][256], bn = b*64+n
         const _Float16* __restrict__ Wih0, const _Float16* __restrict__ Wih1,
         const _Float16* __restrict__ Whh0, const _Float16* __restrict__ Whh1,
         const float* __restrict__ b0, const float* __restrict__ b1,
         float* __restrict__ tag32, _Float16* __restrict__ tag16) {
  __shared__ float    gbuf[16][512];
  __shared__ _Float16 hbuf[16][128];
  const int dir = blockIdx.y;
  const int rowbase = blockIdx.x * 16;               // doc index base
  const _Float16* Wih = dir ? Wih1 : Wih0;
  const _Float16* Whh = dir ? Whh1 : Whh0;
  const float* bias = dir ? b1 : b0;
  const int wave = threadIdx.x >> 5;
  const int tid = threadIdx.x;

  for (int i = tid; i < 16 * 128; i += 128) (&hbuf[0][0])[i] = (_Float16)0.f;
  float c_state[16];
#pragma unroll
  for (int i = 0; i < 16; ++i) c_state[i] = 0.f;
  __syncthreads();

  for (int t = 0; t < 64; ++t) {
    const int pos = dir ? (63 - t) : t;
    v8f acc[8];
#pragma unroll
    for (int n = 0; n < 8; ++n) acc[n] = (v8f){};
    const _Float16* Xp = SP + ((size_t)rowbase * 64 + pos) * 256;
#pragma unroll
    for (int kt = 0; kt < 8; ++kt) {                       // sentpres @ W_ih^T
      v16h a = frag_ld(Xp + kt * 32, 64 * 256);
#pragma unroll
      for (int n = 0; n < 8; ++n) {
        v16h b = frag_ld(Wih + (size_t)(wave * 128 + n * 16) * 256 + kt * 32, 256);
        acc[n] = wmma16(a, b, acc[n]);
      }
    }
#pragma unroll
    for (int kt = 0; kt < 4; ++kt) {                       // h @ W_hh^T
      v16h a = frag_ld(&hbuf[0][kt * 32], 128);
#pragma unroll
      for (int n = 0; n < 8; ++n) {
        v16h b = frag_ld(Whh + (size_t)(wave * 128 + n * 16) * 128 + kt * 32, 128);
        acc[n] = wmma16(a, b, acc[n]);
      }
    }
#pragma unroll
    for (int n = 0; n < 8; ++n) frag_st(&gbuf[0][wave * 128 + n * 16], 512, acc[n]);
    __syncthreads();
#pragma unroll
    for (int r = 0; r < 16; ++r) {
      float gi = gbuf[r][tid]       + bias[tid];
      float gf = gbuf[r][128 + tid] + bias[128 + tid];
      float gg = gbuf[r][256 + tid] + bias[256 + tid];
      float go = gbuf[r][384 + tid] + bias[384 + tid];
      float c = sigmoidf(gf) * c_state[r] + sigmoidf(gi) * tanhf(gg);
      float h = sigmoidf(go) * tanhf(c);
      c_state[r] = c;
      hbuf[r][tid] = (_Float16)h;                          // raw h feeds recurrence
      float ho = tanhf(h);                                 // tanh only on output
      size_t o = (((size_t)(rowbase + r)) * 64 + pos) * 256 + (size_t)dir * 128 + tid;
      tag32[o] = ho;
      tag16[o] = (_Float16)ho;
    }
    __syncthreads();
  }
}

// ---- Generic WMMA GEMM: out = act(A[M][Kp] @ W[Npad][Kp]^T + bias) ----------
__global__ void __launch_bounds__(128)
gemm_wmma(const _Float16* __restrict__ A, int Kp,
          const _Float16* __restrict__ W, const float* __restrict__ bias,
          int Nvalid, int Npad, int act,
          float* __restrict__ out32, _Float16* __restrict__ out16, int ldo) {
  const int rowbase = blockIdx.x * 16;
  const int ntile = blockIdx.y * 4 + (threadIdx.x >> 5);
  if (ntile * 16 >= Npad) return;
  v8f acc = (v8f){};
  const _Float16* Ap = A + (size_t)rowbase * Kp;
  const _Float16* Wp = W + (size_t)ntile * 16 * Kp;
  for (int kt = 0; kt < Kp / 32; ++kt) {
    v16h a = frag_ld(Ap + kt * 32, Kp);
    v16h b = frag_ld(Wp + kt * 32, Kp);
    acc = wmma16(a, b, acc);
  }
  int lane = threadIdx.x & 31;
  int col = lane & 15, rb = (lane >> 4) << 3;
  int gcol = ntile * 16 + col;
  if (gcol >= Nvalid) return;
  float bv = bias ? bias[gcol] : 0.f;
#pragma unroll
  for (int v = 0; v < 8; ++v) {
    float x = acc[v] + bv;
    if (act == 1) x = fmaxf(x, 0.f);
    size_t o = (size_t)(rowbase + rb + v) * ldo + gcol;
    if (out32) out32[o] = x;
    if (out16) out16[o] = (_Float16)x;
  }
}

// ---- SPP: att = q k^T / 16, then pyramid max-pool to 15 features ------------
__global__ void __launch_bounds__(256)
spp_attpool(const _Float16* __restrict__ q_s, const _Float16* __restrict__ k_s,
            const _Float16* __restrict__ q_r, const _Float16* __restrict__ k_r,
            float* __restrict__ sentFt, float* __restrict__ roleFt) {
  __shared__ float att[64][64];
  const int b = blockIdx.x, which = blockIdx.y;
  const _Float16* q = which ? q_r : q_s;
  const _Float16* k = which ? k_r : k_s;
  float* out = which ? roleFt : sentFt;
  for (int idx = threadIdx.x; idx < 64 * 64; idx += 256) {
    int n = idx >> 6, m = idx & 63;
    const _Float16* qp = q + ((size_t)b * 64 + n) * 256;
    const _Float16* kp = k + ((size_t)b * 64 + m) * 256;
    float s = 0.f;
    for (int d = 0; d < 256; ++d) s += (float)qp[d] * (float)kp[d];
    att[n][m] = s * (1.f / 16.f);
  }
  __syncthreads();
  for (int idx = threadIdx.x; idx < 64 * 15; idx += 256) {
    int n = idx / 15, o = idx % 15;
    int s, p;
    if (o == 0)      { s = 1; p = 0; }
    else if (o < 3)  { s = 2; p = o - 1; }
    else if (o < 7)  { s = 4; p = o - 3; }
    else             { s = 8; p = o - 7; }
    int chunk = 64 / s;
    float m = -INFINITY;
    for (int j = p * chunk; j < (p + 1) * chunk; ++j) m = fmaxf(m, att[n][j]);
    out[((size_t)b * 64 + n) * 15 + o] = m;
  }
}

// ---- GCN aggregation: agg = (Cos @ x + 2x) / (N+2), Cos = cosine-sim --------
__global__ void __launch_bounds__(256)
gcn_agg(const float* __restrict__ tag32, _Float16* __restrict__ agg16) {
  __shared__ float Cos[64][64];
  __shared__ float nrm[64];
  const int b = blockIdx.x;
  const float* T = tag32 + (size_t)b * 64 * 256;
  for (int n = threadIdx.x; n < 64; n += 256) {
    float s = 0.f;
    for (int d = 0; d < 256; ++d) { float v = T[n * 256 + d]; s += v * v; }
    nrm[n] = sqrtf(s) + 1e-8f;
  }
  __syncthreads();
  for (int idx = threadIdx.x; idx < 4096; idx += 256) {
    int n = idx >> 6, m = idx & 63;
    float s = 0.f;
    for (int d = 0; d < 256; ++d) s += T[n * 256 + d] * T[m * 256 + d];
    Cos[n][m] = s / (nrm[n] * nrm[m]);
  }
  __syncthreads();
  for (int idx = threadIdx.x; idx < 64 * 256; idx += 256) {
    int n = idx >> 8, d = idx & 255;
    float s = 0.f;
    for (int m = 0; m < 64; ++m) s += Cos[n][m] * T[m * 256 + d];
    s = (s + 2.f * T[n * 256 + d]) * (1.f / 66.f);
    agg16[(size_t)b * 64 * 256 + idx] = (_Float16)s;
  }
}

// ---- feat = [tag_out | sentFt | roleFt | 0-pad], Kp = 288 -------------------
__global__ void build_feat(const _Float16* __restrict__ tag16,
                           const float* __restrict__ sentFt,
                           const float* __restrict__ roleFt,
                           _Float16* __restrict__ feat) {
  size_t total = (size_t)2048 * 288;
  for (size_t i = (size_t)blockIdx.x * blockDim.x + threadIdx.x; i < total;
       i += (size_t)gridDim.x * blockDim.x) {
    size_t r = i / 288; int c = (int)(i % 288);
    float v;
    if (c < 256)      v = (float)tag16[r * 256 + c];
    else if (c < 271) v = sentFt[r * 15 + (c - 256)];
    else if (c < 286) v = roleFt[r * 15 + (c - 271)];
    else              v = 0.f;
    feat[i] = (_Float16)v;
  }
}

__global__ void log_softmax8(const float* __restrict__ logits, float* __restrict__ out) {
  int r = blockIdx.x * blockDim.x + threadIdx.x;
  if (r >= 2048) return;
  const float* p = logits + (size_t)r * 16;
  float mx = p[0];
  for (int c = 1; c < 8; ++c) mx = fmaxf(mx, p[c]);
  float s = 0.f;
  for (int c = 0; c < 8; ++c) s += expf(p[c] - mx);
  float l = logf(s) + mx;
  for (int c = 0; c < 8; ++c) out[(size_t)r * 8 + c] = p[c] - l;
}

// ---- host orchestration -----------------------------------------------------
static inline int pgrid(size_t total) {
  size_t b = (total + 255) / 256;
  return (int)(b > 4096 ? 4096 : b);
}

extern "C" void kernel_launch(void* const* d_in, const int* in_sizes, int n_in,
                              void* d_out, int out_size, void* d_ws, size_t ws_size,
                              hipStream_t stream) {
  (void)in_sizes; (void)n_in; (void)out_size; (void)ws_size;
  const float* documents = (const float*)d_in[0];
  const float* sw_ih_f = (const float*)d_in[1];
  const float* sw_hh_f = (const float*)d_in[2];
  const float* sb_f    = (const float*)d_in[3];
  const float* sw_ih_b = (const float*)d_in[4];
  const float* sw_hh_b = (const float*)d_in[5];
  const float* sb_b    = (const float*)d_in[6];
  const float* sf_Wq = (const float*)d_in[7];
  const float* sf_bq = (const float*)d_in[8];
  const float* sf_Wk = (const float*)d_in[9];
  const float* sf_bk = (const float*)d_in[10];
  const float* rf_Wq = (const float*)d_in[11];
  const float* rf_bq = (const float*)d_in[12];
  const float* rf_Wk = (const float*)d_in[13];
  const float* rf_bk = (const float*)d_in[14];
  const float* tw_ih_f = (const float*)d_in[15];
  const float* tw_hh_f = (const float*)d_in[16];
  const float* tb_f    = (const float*)d_in[17];
  const float* tw_ih_b = (const float*)d_in[18];
  const float* tw_hh_b = (const float*)d_in[19];
  const float* tb_b    = (const float*)d_in[20];
  const float* sage_W = (const float*)d_in[21];
  const float* sage_b = (const float*)d_in[22];
  const float* cls_W  = (const float*)d_in[23];
  const float* cls_b  = (const float*)d_in[24];

  float* result = (float*)d_out;                  // (32,64,8)
  float* gcn_feats = (float*)d_out + 2048 * 8;    // (32,64,256)

  // workspace carve-out
  size_t cur = 0;
  auto alloc = [&](size_t bytes) {
    void* p = (char*)d_ws + cur;
    cur += (bytes + 255) & ~(size_t)255;
    return p;
  };
  _Float16* X16   = (_Float16*)alloc((size_t)102400 * 224 * 2);  // docs, K pad 224
  _Float16* WihS0 = (_Float16*)alloc((size_t)512 * 224 * 2);
  _Float16* WihS1 = (_Float16*)alloc((size_t)512 * 224 * 2);
  _Float16* WhhS0 = (_Float16*)alloc((size_t)512 * 128 * 2);
  _Float16* WhhS1 = (_Float16*)alloc((size_t)512 * 128 * 2);
  _Float16* WihT0 = (_Float16*)alloc((size_t)512 * 256 * 2);
  _Float16* WihT1 = (_Float16*)alloc((size_t)512 * 256 * 2);
  _Float16* WhhT0 = (_Float16*)alloc((size_t)512 * 128 * 2);
  _Float16* WhhT1 = (_Float16*)alloc((size_t)512 * 128 * 2);
  _Float16* WqS = (_Float16*)alloc((size_t)256 * 256 * 2);
  _Float16* WkS = (_Float16*)alloc((size_t)256 * 256 * 2);
  _Float16* WqR = (_Float16*)alloc((size_t)256 * 256 * 2);
  _Float16* WkR = (_Float16*)alloc((size_t)256 * 256 * 2);
  _Float16* SgW = (_Float16*)alloc((size_t)256 * 256 * 2);
  _Float16* ClW = (_Float16*)alloc((size_t)16 * 288 * 2);
  float*    sp32  = (float*)alloc((size_t)2048 * 256 * 4);
  _Float16* sp16  = (_Float16*)alloc((size_t)2048 * 256 * 2);
  float*    tag32 = (float*)alloc((size_t)2048 * 256 * 4);
  _Float16* tag16 = (_Float16*)alloc((size_t)2048 * 256 * 2);
  _Float16* qS16  = (_Float16*)alloc((size_t)2048 * 256 * 2);
  _Float16* kS16  = (_Float16*)alloc((size_t)2048 * 256 * 2);
  _Float16* qR16  = (_Float16*)alloc((size_t)2048 * 256 * 2);
  _Float16* kR16  = (_Float16*)alloc((size_t)2048 * 256 * 2);
  float*    sentFt = (float*)alloc((size_t)2048 * 15 * 4);
  float*    roleFt = (float*)alloc((size_t)2048 * 15 * 4);
  _Float16* agg16  = (_Float16*)alloc((size_t)2048 * 256 * 2);
  _Float16* feat16 = (_Float16*)alloc((size_t)2048 * 288 * 2);
  float*    logits = (float*)alloc((size_t)2048 * 16 * 4);

  // 1) pack inputs & weights to f16 (with K/row zero padding)
  pack_f16<<<pgrid((size_t)102400 * 224), 256, 0, stream>>>(documents, X16, 102400, 102400, 200, 224);
  pack_f16<<<pgrid((size_t)512 * 224), 256, 0, stream>>>(sw_ih_f, WihS0, 512, 512, 200, 224);
  pack_f16<<<pgrid((size_t)512 * 224), 256, 0, stream>>>(sw_ih_b, WihS1, 512, 512, 200, 224);
  pack_f16<<<pgrid((size_t)512 * 128), 256, 0, stream>>>(sw_hh_f, WhhS0, 512, 512, 128, 128);
  pack_f16<<<pgrid((size_t)512 * 128), 256, 0, stream>>>(sw_hh_b, WhhS1, 512, 512, 128, 128);
  pack_f16<<<pgrid((size_t)512 * 256), 256, 0, stream>>>(tw_ih_f, WihT0, 512, 512, 256, 256);
  pack_f16<<<pgrid((size_t)512 * 256), 256, 0, stream>>>(tw_ih_b, WihT1, 512, 512, 256, 256);
  pack_f16<<<pgrid((size_t)512 * 128), 256, 0, stream>>>(tw_hh_f, WhhT0, 512, 512, 128, 128);
  pack_f16<<<pgrid((size_t)512 * 128), 256, 0, stream>>>(tw_hh_b, WhhT1, 512, 512, 128, 128);
  pack_f16<<<pgrid((size_t)256 * 256), 256, 0, stream>>>(sf_Wq, WqS, 256, 256, 256, 256);
  pack_f16<<<pgrid((size_t)256 * 256), 256, 0, stream>>>(sf_Wk, WkS, 256, 256, 256, 256);
  pack_f16<<<pgrid((size_t)256 * 256), 256, 0, stream>>>(rf_Wq, WqR, 256, 256, 256, 256);
  pack_f16<<<pgrid((size_t)256 * 256), 256, 0, stream>>>(rf_Wk, WkR, 256, 256, 256, 256);
  pack_f16<<<pgrid((size_t)256 * 256), 256, 0, stream>>>(sage_W, SgW, 256, 256, 256, 256);
  pack_f16<<<pgrid((size_t)16 * 288), 256, 0, stream>>>(cls_W, ClW, 8, 16, 286, 288);

  // 2) sentence BiLSTM -> sentpres (tanh of time-mean, order-invariant for both dirs)
  sent_lstm<<<dim3(128, 2), 128, 0, stream>>>(X16, WihS0, WihS1, WhhS0, WhhS1,
                                              sb_f, sb_b, sp32, sp16);

  // 3) tag BiLSTM over sentences -> tag_out (tanh)
  tag_lstm<<<dim3(2, 2), 128, 0, stream>>>(sp16, WihT0, WihT1, WhhT0, WhhT1,
                                           tb_f, tb_b, tag32, tag16);

  // 4) SPP q/k projections (WMMA GEMMs), M=2048, N=256, K=256
  gemm_wmma<<<dim3(128, 4), 128, 0, stream>>>(sp16, 256, WqS, sf_bq, 256, 256, 0,
                                              (float*)nullptr, qS16, 256);
  gemm_wmma<<<dim3(128, 4), 128, 0, stream>>>(sp16, 256, WkS, sf_bk, 256, 256, 0,
                                              (float*)nullptr, kS16, 256);
  gemm_wmma<<<dim3(128, 4), 128, 0, stream>>>(tag16, 256, WqR, rf_bq, 256, 256, 0,
                                              (float*)nullptr, qR16, 256);
  gemm_wmma<<<dim3(128, 4), 128, 0, stream>>>(tag16, 256, WkR, rf_bk, 256, 256, 0,
                                              (float*)nullptr, kR16, 256);

  // 5) attention + spatial pyramid max-pool -> sentFt / roleFt
  spp_attpool<<<dim3(32, 2), 256, 0, stream>>>(qS16, kS16, qR16, kR16, sentFt, roleFt);

  // 6) cosine-graph aggregation, then SAGE conv (relu) -> gcn_feats (d_out)
  gcn_agg<<<32, 256, 0, stream>>>(tag32, agg16);
  gemm_wmma<<<dim3(128, 4), 128, 0, stream>>>(agg16, 256, SgW, sage_b, 256, 256, 1,
                                              gcn_feats, (_Float16*)nullptr, 256);

  // 7) feat concat -> classifier -> log_softmax -> result (d_out)
  build_feat<<<pgrid((size_t)2048 * 288), 256, 0, stream>>>(tag16, sentFt, roleFt, feat16);
  gemm_wmma<<<dim3(128, 1), 128, 0, stream>>>(feat16, 288, ClW, cls_b, 8, 16, 0,
                                              logits, (_Float16*)nullptr, 16);
  log_softmax8<<<(2048 + 255) / 256, 256, 0, stream>>>(logits, result);
}
